// DQN_34136400069239
// MI455X (gfx1250) — matrix operand order repaired and checked by reading, not compiled, not run on hardware
//
#include <hip/hip_runtime.h>

typedef __attribute__((ext_vector_type(16))) __bf16 v16bf;
typedef __attribute__((ext_vector_type(8)))  float  v8f;

// ---------- helpers ----------
__device__ __forceinline__ v8f wmma_bf16(v16bf a, v16bf b, v8f c) {
  return __builtin_amdgcn_wmma_f32_16x16x32_bf16(
      false, a, false, b, (short)0, c, false, false);
}

// A-fragment (16x32 bf16) from row-major LDS [m][k], element stride ld
__device__ __forceinline__ v16bf ldsFragA(const char* base, int ld, int kb, int lane) {
  int m = lane & 15, hi = lane >> 4;
  const uint4* p0 = (const uint4*)(base + 2 * (m * ld + kb + hi * 8));
  const uint4* p1 = (const uint4*)(base + 2 * (m * ld + kb + 16 + hi * 8));
  union { uint4 u[2]; v16bf v; } r;
  r.u[0] = *p0; r.u[1] = *p1;
  return r.v;
}

// B-fragment (32x16 bf16) from W^T stored [n][k] in LDS, element stride ld
__device__ __forceinline__ v16bf ldsFragB(const char* base, int ld, int nb, int kb, int lane) {
  int n = lane & 15, hi = lane >> 4;
  const uint4* p0 = (const uint4*)(base + 2 * ((nb + n) * ld + kb + hi * 8));
  const uint4* p1 = (const uint4*)(base + 2 * ((nb + n) * ld + kb + 16 + hi * 8));
  union { uint4 u[2]; v16bf v; } r;
  r.u[0] = *p0; r.u[1] = *p1;
  return r.v;
}

// A-fragment built from global f32 x (row length 64), native cvt to bf16
__device__ __forceinline__ v16bf gblFragA(const float* __restrict__ x,
                                          long long rowBase, int kb, int lane) {
  int m = lane & 15, hi = lane >> 4;
  const float* r = x + (rowBase + m) * 64;
  const float4* q0 = (const float4*)(r + kb + hi * 8);
  const float4* q1 = (const float4*)(r + kb + 16 + hi * 8);
  float4 a = q0[0], b = q0[1], c = q1[0], d = q1[1];
  v16bf v;
  v[0]  = (__bf16)a.x; v[1]  = (__bf16)a.y; v[2]  = (__bf16)a.z; v[3]  = (__bf16)a.w;
  v[4]  = (__bf16)b.x; v[5]  = (__bf16)b.y; v[6]  = (__bf16)b.z; v[7]  = (__bf16)b.w;
  v[8]  = (__bf16)c.x; v[9]  = (__bf16)c.y; v[10] = (__bf16)c.z; v[11] = (__bf16)c.w;
  v[12] = (__bf16)d.x; v[13] = (__bf16)d.y; v[14] = (__bf16)d.z; v[15] = (__bf16)d.w;
  return v;
}

// bias + relu + store f32 16x16 D-tile as bf16 into row-major LDS
__device__ __forceinline__ void biasReluStore(char* base, int ld, int nb, int lane,
                                              v8f c, float bv) {
  int col = lane & 15, hi = lane >> 4;
  __bf16* p = (__bf16*)base;
#pragma unroll
  for (int i = 0; i < 8; ++i)
    p[(i + hi * 8) * ld + nb + col] = (__bf16)fmaxf(c[i] + bv, 0.0f);
}

// ---------- LDS layout (dynamic) ----------
// W1T: [256][72]  bf16  @0       size  36864
// W2T: [256][264] bf16  @36864   size 135168
// W3T: [128][264] bf16  @172032  size  67584
// h1 : per-wave [16][264] bf16 @239616 + wave*8448   (8 waves: 67584)
// h2 : per-wave [16][72]  bf16 @307200 + wave*2304   (8 waves: 18432)
// total 325632 bytes  (<= 320KB WGP LDS)
#define PHI_SMEM 325632

__global__ __launch_bounds__(256)
void phi_kernel(const float* __restrict__ x,
                const float* __restrict__ pw1, const float* __restrict__ pb1,
                const float* __restrict__ pw2, const float* __restrict__ pb2,
                const float* __restrict__ pw3,
                float* __restrict__ fpacc, int ntiles) {
  extern __shared__ char smem[];
  char* W1T = smem;
  char* W2T = smem + 36864;
  char* W3T = smem + 172032;
  const int tid  = threadIdx.x;
  const int wave = tid >> 5;
  const int lane = tid & 31;
  char* h1 = smem + 239616 + wave * 8448;
  char* h2 = smem + 307200 + wave * 2304;

  // stage weights as bf16, transposed [n][k], padded strides
  for (int idx = tid; idx < 64 * 256; idx += 256) {
    int k = idx >> 8, n = idx & 255;
    ((__bf16*)W1T)[n * 72 + k] = (__bf16)pw1[idx];
  }
  for (int idx = tid; idx < 256 * 256; idx += 256) {
    int k = idx >> 8, n = idx & 255;
    ((__bf16*)W2T)[n * 264 + k] = (__bf16)pw2[idx];
  }
  for (int idx = tid; idx < 256 * 128; idx += 256) {
    int k = idx >> 7, n = idx & 127;
    ((__bf16*)W3T)[n * 264 + k] = (__bf16)pw3[idx];
  }
  __syncthreads();

  // persistent layer-3 accumulators: C-accumulation performs the sum-pool
  v8f acc[8];
#pragma unroll
  for (int t = 0; t < 8; ++t) acc[t] = (v8f)0.0f;

  const int gw = blockIdx.x * 8 + wave;
  const int nw = gridDim.x * 8;
  for (int tile = gw; tile < ntiles; tile += nw) {
    long long rowBase = (long long)tile * 16;

    // ---- layer 1: [16,64] x [64,256], output tiles in independent pairs ----
    v16bf a0 = gblFragA(x, rowBase, 0, lane);
    v16bf a1 = gblFragA(x, rowBase, 32, lane);
#pragma unroll
    for (int jp = 0; jp < 8; ++jp) {
      int nb0 = (2 * jp) * 16, nb1 = nb0 + 16;
      v16bf b00 = ldsFragB(W1T, 72, nb0, 0,  lane);
      v16bf b10 = ldsFragB(W1T, 72, nb1, 0,  lane);
      v16bf b01 = ldsFragB(W1T, 72, nb0, 32, lane);
      v16bf b11 = ldsFragB(W1T, 72, nb1, 32, lane);
      v8f c0 = (v8f)0.0f, c1 = (v8f)0.0f;
      c0 = wmma_bf16(a0, b00, c0);
      c1 = wmma_bf16(a0, b10, c1);
      c0 = wmma_bf16(a1, b01, c0);
      c1 = wmma_bf16(a1, b11, c1);
      biasReluStore(h1, 264, nb0, lane, c0, pb1[nb0 + (lane & 15)]);
      biasReluStore(h1, 264, nb1, lane, c1, pb1[nb1 + (lane & 15)]);
    }

    // cache all layer-2 A-fragments (h1) in registers
    v16bf Af[8];
#pragma unroll
    for (int ks = 0; ks < 8; ++ks) Af[ks] = ldsFragA(h1, 264, ks * 32, lane);

    // ---- layer 2 + fused layer 3, in K-quarters ----
#pragma unroll 1
    for (int q = 0; q < 4; ++q) {
      // layer 2: two independent accumulator chains, pipelined B loads
#pragma unroll
      for (int jp = 0; jp < 2; ++jp) {
        int nb0 = (q * 4 + 2 * jp) * 16, nb1 = nb0 + 16;
        v16bf b0 = ldsFragB(W2T, 264, nb0, 0, lane);
        v16bf b1 = ldsFragB(W2T, 264, nb1, 0, lane);
        v8f c0 = (v8f)0.0f, c1 = (v8f)0.0f;
#pragma unroll
        for (int ks = 0; ks < 8; ++ks) {
          v16bf n0 = b0, n1 = b1;
          if (ks < 7) {
            n0 = ldsFragB(W2T, 264, nb0, (ks + 1) * 32, lane);
            n1 = ldsFragB(W2T, 264, nb1, (ks + 1) * 32, lane);
          }
          c0 = wmma_bf16(Af[ks], b0, c0);
          c1 = wmma_bf16(Af[ks], b1, c1);
          b0 = n0; b1 = n1;
        }
        biasReluStore(h2, 72, (2 * jp) * 16,     lane, c0, pb2[nb0 + (lane & 15)]);
        biasReluStore(h2, 72, (2 * jp) * 16 + 16, lane, c1, pb2[nb1 + (lane & 15)]);
      }
      // layer 3: 8 independent accumulator chains, pipelined B loads
#pragma unroll
      for (int k3 = 0; k3 < 2; ++k3) {
        v16bf a3 = ldsFragA(h2, 72, k3 * 32, lane);
        v16bf bt = ldsFragB(W3T, 264, 0, q * 64 + k3 * 32, lane);
#pragma unroll
        for (int t = 0; t < 8; ++t) {
          v16bf bn = bt;
          if (t < 7) bn = ldsFragB(W3T, 264, (t + 1) * 16, q * 64 + k3 * 32, lane);
          acc[t] = wmma_bf16(a3, bt, acc[t]);
          bt = bn;
        }
      }
    }
  }

  // reduce 16 rows of each accumulated 16x16 tile -> column sums -> global
#pragma unroll
  for (int t = 0; t < 8; ++t) {
    float s = 0.0f;
#pragma unroll
    for (int i = 0; i < 8; ++i) s += acc[t][i];
    s += __shfl_xor(s, 16);
    if (lane < 16) atomicAdd(&fpacc[t * 16 + lane], s);
  }
}

__global__ void zero_kernel(float* p, int n) {
  int i = blockIdx.x * blockDim.x + threadIdx.x;
  if (i < n) p[i] = 0.0f;
}

// rho network + head: tiny, one block of 256 threads
__global__ void rho_kernel(const float* __restrict__ fpacc, float nrows,
                           const float* __restrict__ pb3,
                           const float* __restrict__ rw1, const float* __restrict__ rb1,
                           const float* __restrict__ rw2, const float* __restrict__ rb2,
                           const float* __restrict__ rw3, const float* __restrict__ rb3,
                           const float* __restrict__ xs,
                           const float* __restrict__ w1, const float* __restrict__ b1,
                           const float* __restrict__ w2, const float* __restrict__ b2,
                           const float* __restrict__ w3, const float* __restrict__ b3,
                           float* __restrict__ out) {
  __shared__ float v0[256], v1[256], cat[160];
  int j = threadIdx.x;
  if (j < 128) v0[j] = fpacc[j] + nrows * pb3[j];   // fp = sum(phi) (+N*pb3)
  __syncthreads();
  { float s = rb1[j]; for (int i = 0; i < 128; ++i) s += v0[i] * rw1[i * 256 + j];
    v1[j] = fmaxf(s, 0.0f); }
  __syncthreads();
  { float s = rb2[j]; for (int i = 0; i < 256; ++i) s += v1[i] * rw2[i * 256 + j];
    v0[j] = fmaxf(s, 0.0f); }
  __syncthreads();
  if (j < 128) { float s = rb3[j]; for (int i = 0; i < 256; ++i) s += v0[i] * rw3[i * 128 + j];
    cat[j] = s; }
  if (j < 16) cat[128 + j] = xs[j];
  __syncthreads();
  { float s = b1[j]; for (int i = 0; i < 144; ++i) s += cat[i] * w1[i * 256 + j];
    v1[j] = fmaxf(s, 0.0f); }
  __syncthreads();
  { float s = b2[j]; for (int i = 0; i < 256; ++i) s += v1[i] * w2[i * 256 + j];
    v0[j] = fmaxf(s, 0.0f); }
  __syncthreads();
  if (j < 5) { float s = b3[j]; for (int i = 0; i < 256; ++i) s += v0[i] * w3[i * 5 + j];
    out[j] = s; }
}

extern "C" void kernel_launch(void* const* d_in, const int* in_sizes, int n_in,
                              void* d_out, int out_size, void* d_ws, size_t ws_size,
                              hipStream_t stream) {
  const float* x   = (const float*)d_in[0];
  const float* xs  = (const float*)d_in[1];
  const float* pw1 = (const float*)d_in[2];
  const float* pb1 = (const float*)d_in[3];
  const float* pw2 = (const float*)d_in[4];
  const float* pb2 = (const float*)d_in[5];
  const float* pw3 = (const float*)d_in[6];
  const float* pb3 = (const float*)d_in[7];
  const float* rw1 = (const float*)d_in[8];
  const float* rb1 = (const float*)d_in[9];
  const float* rw2 = (const float*)d_in[10];
  const float* rb2 = (const float*)d_in[11];
  const float* rw3 = (const float*)d_in[12];
  const float* rb3 = (const float*)d_in[13];
  const float* w1  = (const float*)d_in[14];
  const float* b1  = (const float*)d_in[15];
  const float* w2  = (const float*)d_in[16];
  const float* b2  = (const float*)d_in[17];
  const float* w3  = (const float*)d_in[18];
  const float* b3  = (const float*)d_in[19];

  float* fpacc = (float*)d_ws;                 // 128 f32 partial feature sums
  int nrows  = in_sizes[0] / 64;
  int ntiles = nrows / 16;

  zero_kernel<<<1, 128, 0, stream>>>(fpacc, 128);
  phi_kernel<<<dim3(512), dim3(256), PHI_SMEM, stream>>>(
      x, pw1, pb1, pw2, pb2, pw3, fpacc, ntiles);
  rho_kernel<<<1, 256, 0, stream>>>(fpacc, (float)nrows, pb3,
      rw1, rb1, rw2, rb2, rw3, rb3, xs, w1, b1, w2, b2, w3, b3,
      (float*)d_out);
}
